// EncoderBase_10411000726279
// MI455X (gfx1250) — matrix operand order repaired and checked by reading, not compiled, not run on hardware
//
#include <hip/hip_runtime.h>

#define DZ   2
#define DKK  2048
#define DIN  1024
#define NH   16
#define DEL  64
#define OUTD 3072   // 3 * NH * DEL

typedef __attribute__((ext_vector_type(16))) __bf16 v16bf;
typedef __attribute__((ext_vector_type(8)))  float  v8f;

struct U32x8 { uint4 a, b; };
struct US16  { unsigned short u[16]; };

static __device__ __forceinline__ v16bf ld8x2(const unsigned short* p0, const unsigned short* p1) {
    U32x8 t;
    t.a = *reinterpret_cast<const uint4*>(p0);
    t.b = *reinterpret_cast<const uint4*>(p1);
    return __builtin_bit_cast(v16bf, t);
}
static __device__ __forceinline__ v16bf ld16c(const unsigned short* p) { return ld8x2(p, p + 8); }

static __device__ __forceinline__ unsigned short f2bf(float f) {
    unsigned u = __float_as_uint(f);
    return (unsigned short)((u + 0x7FFFu + ((u >> 16) & 1u)) >> 16);   // RNE
}
static __device__ __forceinline__ v16bf packbf16(const float* f) {
    US16 t;
#pragma unroll
    for (int i = 0; i < 16; ++i) t.u[i] = f2bf(f[i]);
    return __builtin_bit_cast(v16bf, t);
}

static __device__ __forceinline__ v8f wmma_bf16(v16bf a, v16bf b, v8f c) {
    // D = A(16x32 bf16) * B(32x16 bf16) + C(16x16 f32)
    return __builtin_amdgcn_wmma_f32_16x16x32_bf16(false, a, false, b, (short)0, c, false, false);
}

// ---------------------------------------------------------------------------
// Kernel 1: f32 -> bf16 conversion of x and W into workspace
// ---------------------------------------------------------------------------
__global__ __launch_bounds__(256) void cvt_kernel(const float* __restrict__ x,
                                                  const float* __restrict__ W,
                                                  unsigned short* __restrict__ xb,
                                                  unsigned short* __restrict__ wb) {
    const size_t NX = (size_t)DZ * DKK * DIN;
    const size_t NW = (size_t)OUTD * DIN;
    size_t i = ((size_t)blockIdx.x * blockDim.x + threadIdx.x) * 4;
    if (i < NX) {
        float4 f = *reinterpret_cast<const float4*>(x + i);
        ushort4 u; u.x = f2bf(f.x); u.y = f2bf(f.y); u.z = f2bf(f.z); u.w = f2bf(f.w);
        *reinterpret_cast<ushort4*>(xb + i) = u;
    } else if (i < NX + NW) {
        size_t j = i - NX;
        float4 f = *reinterpret_cast<const float4*>(W + j);
        ushort4 u; u.x = f2bf(f.x); u.y = f2bf(f.y); u.z = f2bf(f.z); u.w = f2bf(f.w);
        *reinterpret_cast<ushort4*>(wb + j) = u;
    }
}

// ---------------------------------------------------------------------------
// Kernel 2: QKV GEMM, ping-pong double-buffered. One wave -> 32(m) x 64(n).
// ---------------------------------------------------------------------------
static __device__ __forceinline__ void qkv_step(
    const unsigned short* __restrict__ arow0, const unsigned short* __restrict__ arow1,
    const unsigned short* __restrict__ wb, int n0, int c16, int half, int b8, int kn,
    v16bf& a0c, v16bf& a1c, v16bf* bbc,          // compute set (current K-step)
    v16bf& a0n, v16bf& a1n, v16bf* bbn,          // load set (next K-step)
    v8f (*acc)[4])
{
    // prefetch next K-step into the other buffer set (issued before the WMMAs)
    a0n = ld16c(arow0 + kn + half * 16);
    a1n = ld16c(arow1 + kn + half * 16);
#pragma unroll
    for (int t = 0; t < 4; ++t)
        bbn[t] = ld16c(wb + (long long)(n0 + t * 16 + c16) * DIN + kn + half * 16);
#pragma unroll
    for (int t = 0; t < 4; ++t) {
        acc[0][t] = wmma_bf16(a0c, bbc[t], acc[0][t]);
        acc[1][t] = wmma_bf16(a1c, bbc[t], acc[1][t]);
    }
}

__global__ __launch_bounds__(256) void qkv_kernel(
    const unsigned short* __restrict__ xb, const unsigned short* __restrict__ wb,
    const float* __restrict__ bias, const float* __restrict__ cosp, const float* __restrict__ sinp,
    unsigned short* __restrict__ qb, unsigned short* __restrict__ kbuf,
    unsigned short* __restrict__ vtb)
{
    __shared__ float tl[8][16][65];                 // wave-private 16x64 tile (padded)
    const int lane = threadIdx.x & 31;
    const int wv   = threadIdx.x >> 5;
    const int wid  = blockIdx.x * 8 + wv;
    const int mt = wid / (OUTD / 64), nt = wid % (OUTD / 64);
    const int m0 = mt * 32, n0 = nt * 64;
    const int c16 = lane & 15, half = lane >> 4, b8 = half * 8;

    v8f acc[2][4];
#pragma unroll
    for (int s = 0; s < 2; ++s)
#pragma unroll
        for (int t = 0; t < 4; ++t)
#pragma unroll
            for (int i = 0; i < 8; ++i) acc[s][t][i] = 0.f;

    const unsigned short* arow0 = xb + (long long)(m0 + c16) * DIN;
    const unsigned short* arow1 = arow0 + 16 * DIN;

    // prologue: set A <- K-step 0
    v16bf a0A = ld16c(arow0 + half * 16);
    v16bf a1A = ld16c(arow1 + half * 16);
    v16bf bbA[4];
#pragma unroll
    for (int t = 0; t < 4; ++t)
        bbA[t] = ld16c(wb + (long long)(n0 + t * 16 + c16) * DIN + half * 16);
    v16bf a0B, a1B, bbB[4];

    for (int kk = 0; kk < DIN; kk += 64) {
        qkv_step(arow0, arow1, wb, n0, c16, half, b8, kk + 32,
                 a0A, a1A, bbA, a0B, a1B, bbB, acc);           // compute A, load B
        qkv_step(arow0, arow1, wb, n0, c16, half, b8, (kk + 64) & (DIN - 1),
                 a0B, a1B, bbB, a0A, a1A, bbA, acc);           // compute B, load A
    }

    const int sel = n0 / 1024;
    const int h   = (n0 % 1024) / 64;
    const int z   = m0 / DKK;                 // 32-row tile never crosses z (2048 % 32 == 0)
    const int row = c16;

#pragma unroll
    for (int sub = 0; sub < 2; ++sub) {
        // bias + park this 16x64 subtile in LDS [row][col] (wave-private, DS in-order)
#pragma unroll
        for (int t = 0; t < 4; ++t) {
            float bv = bias[n0 + t * 16 + c16];
#pragma unroll
            for (int i = 0; i < 8; ++i)
                tl[wv][i + half * 8][t * 16 + c16] = acc[sub][t][i] + bv;
        }
        const int pos = (m0 % DKK) + sub * 16 + row;

        if (sel < 2) {  // RoPE; half 0 -> e<32, half 1 -> e>=32
            unsigned short* ob =
                (sel == 0 ? qb : kbuf) + (((long long)z * NH + h) * DKK + pos) * DEL;
            for (int e2 = 0; e2 < 32; ++e2) {
                float t1 = tl[wv][row][e2];
                float t2 = tl[wv][row][e2 + 32];
                float c = cosp[pos * 32 + e2], s = sinp[pos * 32 + e2];
                float o = (half == 0) ? (t1 * c - t2 * s) : (t2 * c + t1 * s);
                ob[half * 32 + e2] = f2bf(o);
            }
        } else {        // v: cube + normalize by (max|v|)^3 ; store transposed [e][t]
            float m = 0.f;
            for (int j = 0; j < 32; ++j) m = fmaxf(m, fabsf(tl[wv][row][half * 32 + j]));
            m = fmaxf(m, __shfl_xor(m, 16, 32));
            float inv = 1.f / fmaxf(m * m * m, 1e-38f);
            unsigned short* ob = vtb + ((long long)z * NH + h) * DEL * DKK;
            for (int j = 0; j < 32; ++j) {
                float t = tl[wv][row][half * 32 + j];
                ob[(long long)(half * 32 + j) * DKK + pos] = f2bf(t * t * t * inv);
            }
        }
    }
}

// ---------------------------------------------------------------------------
// Kernel 3: attention, single pass (normalizer is a plain sum -> divide once).
// One wave per (z, h, 16-row q-tile); k in chunks of 32, ping-pong K buffers.
// ---------------------------------------------------------------------------
static __device__ __forceinline__ void attn_chunk(
    int kk, int kn,
    const unsigned short* __restrict__ kbase, const unsigned short* __restrict__ vbase,
    const unsigned char* __restrict__ mrow,
    int c16, int half, int b8, unsigned qm,
    v16bf qa0, v16bf qa1,
    v16bf (*kcur)[2], unsigned* mkcur,           // fragments for THIS chunk
    v16bf (*knxt)[2], unsigned* mknxt,           // loaded here for NEXT chunk
    v8f* oacc, float* rs, float (*sl)[33])
{
    // V fragments for THIS chunk: independent of scores, issue first
    v16bf vb[4];
#pragma unroll
    for (int a = 0; a < 4; ++a)
        vb[a] = ld16c(vbase + (long long)(a * 16 + c16) * DKK + kk + half * 16);
    // prefetch NEXT chunk's K fragments + mask bytes into the other set
#pragma unroll
    for (int t = 0; t < 2; ++t) {
        const unsigned short* kr = kbase + (long long)(kn + t * 16 + c16) * DEL + half * 16;
        knxt[t][0] = ld16c(kr);
        knxt[t][1] = ld16c(kr + 32);
        mknxt[t] = mrow[kn + t * 16 + c16];
    }
    // warm L2 one more chunk ahead (global_prefetch_b8)
    __builtin_prefetch(kbase + (long long)(((kk + 64) & (DKK - 1)) + c16) * DEL, 0, 3);
    __builtin_prefetch(vbase + (long long)c16 * DKK + ((kk + 64) & (DKK - 1)), 0, 3);

    // scores: W = Q * K^T (K = 64 in two x32 steps), mask, square, row-sum
#pragma unroll
    for (int t = 0; t < 2; ++t) {
        v8f sa;
#pragma unroll
        for (int i = 0; i < 8; ++i) sa[i] = 0.f;
        sa = wmma_bf16(qa0, kcur[t][0], sa);
        sa = wmma_bf16(qa1, kcur[t][1], sa);
        const unsigned mk = mkcur[t];
#pragma unroll
        for (int i = 0; i < 8; ++i) {
            const int row = i + half * 8;
            float w  = (mk | ((qm >> row) & 1u)) ? 0.f : sa[i];
            float sv = w * w;
            rs[i] += sv;
            sl[row][t * 16 + c16] = sv;              // accum-layout -> [row][col]
        }
    }
    // re-read S (16x32) in A-operand layout, convert to bf16 (DS in-order per wave)
    float sf[16];
#pragma unroll
    for (int j = 0; j < 8; ++j) {
        sf[j]     = sl[c16][half * 16 + j];
        sf[8 + j] = sl[c16][half * 16 + 8 + j];
    }
    v16bf sA = packbf16(sf);
#pragma unroll
    for (int a = 0; a < 4; ++a)
        oacc[a] = wmma_bf16(sA, vb[a], oacc[a]);     // O += S * V
}

__global__ __launch_bounds__(256) void attn_kernel(
    const unsigned short* __restrict__ qb, const unsigned short* __restrict__ kbuf,
    const unsigned short* __restrict__ vtb, const unsigned char* __restrict__ maskp,
    float* __restrict__ out)
{
    __shared__ float sl[8][16][33];                 // wave-private 16x32 S tile (padded)
    const int lane = threadIdx.x & 31;
    const int wv   = threadIdx.x >> 5;
    const int wid  = blockIdx.x * 8 + wv;
    const int z   = wid / (NH * (DKK / 16));
    const int rem = wid % (NH * (DKK / 16));
    const int h   = rem / (DKK / 16);
    const int q0  = (rem % (DKK / 16)) * 16;
    const int c16 = lane & 15, half = lane >> 4, b8 = half * 8;

    unsigned qm = 0;
    for (int i = 0; i < 16; ++i) qm |= (maskp[z * DKK + q0 + i] ? 1u : 0u) << i;

    const unsigned short* qrow = qb + (((long long)z * NH + h) * DKK + q0 + c16) * DEL;
    v16bf qa0 = ld16c(qrow + half * 16);                     // e 0..31
    v16bf qa1 = ld16c(qrow + 32 + half * 16);                // e 32..63

    const unsigned short* kbase = kbuf + ((long long)z * NH + h) * DKK * DEL;
    const unsigned short* vbase = vtb  + ((long long)z * NH + h) * DEL * DKK;
    const unsigned char*  mrow  = maskp + z * DKK;

    v8f oacc[4];
#pragma unroll
    for (int a = 0; a < 4; ++a)
#pragma unroll
        for (int i = 0; i < 8; ++i) oacc[a][i] = 0.f;
    float rs[8] = {};

    // prologue: set A <- chunk 0
    v16bf kcA[2][2], kcB[2][2];
    unsigned mkA[2], mkB[2];
#pragma unroll
    for (int t = 0; t < 2; ++t) {
        const unsigned short* kr = kbase + (long long)(t * 16 + c16) * DEL + half * 16;
        kcA[t][0] = ld16c(kr);
        kcA[t][1] = ld16c(kr + 32);
        mkA[t] = mrow[t * 16 + c16];
    }

    for (int kk = 0; kk < DKK; kk += 64) {
        attn_chunk(kk, kk + 32, kbase, vbase, mrow, c16, half, b8, qm, qa0, qa1,
                   kcA, mkA, kcB, mkB, oacc, rs, sl[wv]);      // compute A, load B
        attn_chunk(kk + 32, (kk + 64) & (DKK - 1), kbase, vbase, mrow, c16, half, b8, qm, qa0, qa1,
                   kcB, mkB, kcA, mkA, oacc, rs, sl[wv]);      // compute B, load A
    }

    // reduce row sums across the 16 lanes holding each row's columns
#pragma unroll
    for (int i = 0; i < 8; ++i) {
        float v = rs[i];
        v += __shfl_xor(v, 1, 32);
        v += __shfl_xor(v, 2, 32);
        v += __shfl_xor(v, 4, 32);
        v += __shfl_xor(v, 8, 32);
        rs[i] = 1.f / fmaxf(v, 1e-38f);
    }
#pragma unroll
    for (int a = 0; a < 4; ++a)
#pragma unroll
        for (int i = 0; i < 8; ++i)
            out[((long long)z * DKK + q0 + i + half * 8) * (NH * DEL) + h * DEL + a * 16 + c16] =
                oacc[a][i] * rs[i];
}

// ---------------------------------------------------------------------------
extern "C" void kernel_launch(void* const* d_in, const int* in_sizes, int n_in,
                              void* d_out, int out_size, void* d_ws, size_t ws_size,
                              hipStream_t stream) {
    const float* x    = (const float*)d_in[0];
    const float* W    = (const float*)d_in[1];
    const float* b    = (const float*)d_in[2];
    const float* cosp = (const float*)d_in[3];
    const float* sinp = (const float*)d_in[4];
    const unsigned char* mask = (const unsigned char*)d_in[5];  // jax bool -> bytes
    float* out = (float*)d_out;

    const size_t NX = (size_t)DZ * DKK * DIN;    // 4,194,304
    const size_t NW = (size_t)OUTD * DIN;        // 3,145,728
    const size_t NQ = (size_t)DZ * NH * DKK * DEL;

    char* ws = (char*)d_ws;
    unsigned short* xb  = (unsigned short*)ws;  ws += NX * 2;
    unsigned short* wb  = (unsigned short*)ws;  ws += NW * 2;
    unsigned short* qb  = (unsigned short*)ws;  ws += NQ * 2;
    unsigned short* kb  = (unsigned short*)ws;  ws += NQ * 2;
    unsigned short* vtb = (unsigned short*)ws;  ws += NQ * 2;   // ~38 MB total

    const int cvtBlocks  = (int)((NX + NW) / 4 / 256);                      // 7168
    const int qkvBlocks  = ((DZ * DKK / 32) * (OUTD / 64)) / 8;             // 768
    const int attnBlocks = (DZ * NH * (DKK / 16)) / 8;                      // 512

    cvt_kernel <<<cvtBlocks, 256, 0, stream>>>(x, W, xb, wb);
    qkv_kernel <<<qkvBlocks, 256, 0, stream>>>(xb, wb, b, cosp, sinp, qb, kb, vtb);
    attn_kernel<<<attnBlocks, 256, 0, stream>>>(qb, kb, vtb, mask, out);
}
